// MultiHeadAttention_54692113547384
// MI455X (gfx1250) — matrix operand order repaired and checked by reading, compiled, not run on hardware
//
#include <hip/hip_runtime.h>
#include <hip/hip_bf16.h>

#define D_MODEL 1024
#define N_HEADS 16
#define DK      64
#define BATCH   4
#define SEQ     2048
#define M_TOK   (BATCH * SEQ)   // 8192
#define LN_EPS  1e-5f

#define TK 32
#define AST 40    // LDS stride (halves) for 32-wide K tiles (pad keeps 16B align)
#define SST 72    // LDS stride for 64-wide K tiles

typedef __bf16 bf16_t;
typedef bf16_t v16bf  __attribute__((ext_vector_type(16)));
typedef bf16_t bf16x2 __attribute__((ext_vector_type(2)));
typedef float  v8f    __attribute__((ext_vector_type(8)));
typedef float  f32x2  __attribute__((ext_vector_type(2)));

union FragU { uint4 q[2]; v16bf v; };

__device__ __forceinline__ unsigned short bf16_bits(float f) {
  __bf16 h = (__bf16)f;
  return __builtin_bit_cast(unsigned short, h);
}
// packed f32x2 -> bf16x2 : lowers to a single v_cvt_pk_bf16_f32
__device__ __forceinline__ unsigned int pack2_bf16(float a, float b) {
  f32x2 f; f.x = a; f.y = b;
  bf16x2 h = __builtin_convertvector(f, bf16x2);
  return __builtin_bit_cast(unsigned int, h);
}

// ---- CDNA5 async copy global->LDS (GV mode), tracked by ASYNCcnt ----
__device__ __forceinline__ void async_ld_b128(const unsigned short* ldsDst, const void* gsrc) {
  unsigned lds = (unsigned)(unsigned long long)ldsDst;   // low 32 bits = LDS offset
  unsigned long long ga = (unsigned long long)gsrc;
  asm volatile("global_load_async_to_lds_b128 %0, %1, off"
               :: "v"(lds), "v"(ga) : "memory");
}
__device__ __forceinline__ void wait_async() {
  asm volatile("s_wait_asynccnt 0x0" ::: "memory");
}

// A fragment (16x32 bf16): lane l holds row m0+(l&15); element i -> k = (i&7)+((i&8)<<1)+((l&16)>>1)
__device__ __forceinline__ v16bf load_A_frag(const unsigned short* tile, int m0, int k0, int stride) {
  const int lane = threadIdx.x & 31;
  const unsigned short* p = tile + (m0 + (lane & 15)) * stride + k0 + ((lane & 16) >> 1);
  FragU f;
  f.q[0] = *(const uint4*)p;          // k .. k+7
  f.q[1] = *(const uint4*)(p + 16);   // k+16 .. k+23
  return f.v;
}
// B fragment (32x16 bf16), tile stored [N][K]: lane l holds col n0+(l&15); element i -> k = k0+(i&7)+((i&8)<<1)+((l&16)>>1)
__device__ __forceinline__ v16bf load_B_frag(const unsigned short* tile, int n0, int k0, int stride) {
  const int lane = threadIdx.x & 31;
  const unsigned short* p = tile + (n0 + (lane & 15)) * stride + k0 + ((lane & 16) >> 1);
  FragU f;
  f.q[0] = *(const uint4*)p;
  f.q[1] = *(const uint4*)(p + 16);
  return f.v;
}
__device__ __forceinline__ v8f wmma_bf16(v16bf a, v16bf b, v8f c) {
  return __builtin_amdgcn_wmma_f32_16x16x32_bf16(false, a, false, b, (short)0, c, false, false);
}

// ---------------- weight f32 -> bf16 ----------------
__global__ __launch_bounds__(256) void convert_bf16_kernel(const float* __restrict__ src,
                                                           unsigned short* __restrict__ dst, int n) {
  int i = blockIdx.x * 256 + threadIdx.x;
  int stride = gridDim.x * 256;
  for (; i < n; i += stride) dst[i] = bf16_bits(src[i]);
}

// ---------------- QKV projection: Hout[b,h,s,d] = bf16(X @ W^T + bias) ----------------
// 64x64 block tile, 8 waves, 2 accumulators/wave.
__global__ __launch_bounds__(256) void proj_qkv_kernel(const float* __restrict__ X,
                                                       const unsigned short* __restrict__ Wb,
                                                       const float* __restrict__ bias,
                                                       unsigned short* __restrict__ Hout) {
  __shared__ unsigned short At[64 * AST];
  __shared__ unsigned short Bt[64 * AST];
  const int tid = threadIdx.x, lane = tid & 31, wave = tid >> 5;
  const int wm = wave >> 1, wn = wave & 1;
  const int mBase = blockIdx.y * 64, nBase = blockIdx.x * 64;
  const int arow = tid >> 2, aseg = (tid & 3) << 3;   // 64 rows x 8 halves
  v8f acc0 = {}, acc1 = {};

  for (int k0 = 0; k0 < D_MODEL; k0 += TK) {
    const float4* asrc = (const float4*)(X + (size_t)(mBase + arow) * D_MODEL + k0 + aseg);
    if (k0 + TK < D_MODEL) __builtin_prefetch((const char*)asrc + TK * 4, 0, 1);
    float4 f0 = asrc[0], f1 = asrc[1];
    uint4 pk;
    pk.x = pack2_bf16(f0.x, f0.y); pk.y = pack2_bf16(f0.z, f0.w);
    pk.z = pack2_bf16(f1.x, f1.y); pk.w = pack2_bf16(f1.z, f1.w);
    *(uint4*)(At + arow * AST + aseg) = pk;
    // weights copied verbatim via async DMA to LDS
    async_ld_b128(Bt + arow * AST + aseg,
                  Wb + (size_t)(nBase + arow) * D_MODEL + k0 + aseg);
    wait_async();
    __syncthreads();
    v16bf a  = load_A_frag(At, wm * 16, 0, AST);
    v16bf b0 = load_B_frag(Bt, wn * 32, 0, AST);
    v16bf b1 = load_B_frag(Bt, wn * 32 + 16, 0, AST);
    acc0 = wmma_bf16(a, b0, acc0);
    acc1 = wmma_bf16(a, b1, acc1);
    __syncthreads();
  }

  const int mLoc = wm * 16 + ((lane >> 4) & 1) * 8;
  const int nCol0 = nBase + wn * 32 + (lane & 15);
  const int nCol1 = nCol0 + 16;
  const float bv0 = bias[nCol0], bv1 = bias[nCol1];
  const int h0 = nCol0 >> 6, d0 = nCol0 & 63;
  const int h1 = nCol1 >> 6, d1 = nCol1 & 63;
#pragma unroll
  for (int r = 0; r < 8; ++r) {
    int m = mBase + mLoc + r;
    int bb = m >> 11, s = m & (SEQ - 1);
    size_t rowb = (size_t)bb * N_HEADS;
    Hout[((rowb + h0) * SEQ + s) * DK + d0] = bf16_bits(acc0[r] + bv0);
    Hout[((rowb + h1) * SEQ + s) * DK + d1] = bf16_bits(acc1[r] + bv1);
  }
}

// ---------------- scores: attn_raw[bh,q,k] = (Qh . Kh) / 8, 64x64 tiles ----------------
__global__ __launch_bounds__(256) void scores_kernel(const unsigned short* __restrict__ Qh,
                                                     const unsigned short* __restrict__ Kh,
                                                     float* __restrict__ attn) {
  __shared__ unsigned short At[64 * SST];
  __shared__ unsigned short Bt[64 * SST];
  const int tid = threadIdx.x, lane = tid & 31, wave = tid >> 5;
  const int wm = wave >> 1, wn = wave & 1;
  const int bh = blockIdx.z;
  const int qBase = blockIdx.y * 64, kBase = blockIdx.x * 64;
  {
    int row = tid >> 2, seg = (tid & 3) << 4;  // 16 halves each
    const unsigned short* qsrc = Qh + (((size_t)bh * SEQ) + qBase + row) * DK + seg;
    const unsigned short* ksrc = Kh + (((size_t)bh * SEQ) + kBase + row) * DK + seg;
    async_ld_b128(At + row * SST + seg,     qsrc);
    async_ld_b128(At + row * SST + seg + 8, qsrc + 8);
    async_ld_b128(Bt + row * SST + seg,     ksrc);
    async_ld_b128(Bt + row * SST + seg + 8, ksrc + 8);
  }
  wait_async();
  __syncthreads();

  v8f acc0 = {}, acc1 = {};
#pragma unroll
  for (int k0 = 0; k0 < DK; k0 += 32) {
    v16bf a  = load_A_frag(At, wm * 16, k0, SST);
    v16bf b0 = load_B_frag(Bt, wn * 32, k0, SST);
    v16bf b1 = load_B_frag(Bt, wn * 32 + 16, k0, SST);
    acc0 = wmma_bf16(a, b0, acc0);
    acc1 = wmma_bf16(a, b1, acc1);
  }
  const int mLoc = wm * 16 + ((lane >> 4) & 1) * 8;
  const int nCol = kBase + wn * 32 + (lane & 15);
  float* out = attn + ((size_t)bh * SEQ + qBase + mLoc) * SEQ + nCol;
#pragma unroll
  for (int r = 0; r < 8; ++r) {
    out[(size_t)r * SEQ]      = acc0[r] * 0.125f;
    out[(size_t)r * SEQ + 16] = acc1[r] * 0.125f;
  }
}

// ---------------- softmax over rows of 2048, in place ----------------
__global__ __launch_bounds__(256) void softmax_kernel(float* __restrict__ attn) {
  __shared__ float red[8];
  float* p = attn + (size_t)blockIdx.x * SEQ;
  const int tid = threadIdx.x;
  float x[8];
#pragma unroll
  for (int i = 0; i < 8; ++i) x[i] = p[tid + i * 256];
  float m = x[0];
#pragma unroll
  for (int i = 1; i < 8; ++i) m = fmaxf(m, x[i]);
  for (int off = 16; off > 0; off >>= 1) m = fmaxf(m, __shfl_down(m, off, 32));
  if ((tid & 31) == 0) red[tid >> 5] = m;
  __syncthreads();
  if (tid < 32) {
    float t = (tid < 8) ? red[tid] : -3.4e38f;
    for (int off = 4; off > 0; off >>= 1) t = fmaxf(t, __shfl_down(t, off, 32));
    if (tid == 0) red[0] = t;
  }
  __syncthreads();
  m = red[0];
  __syncthreads();
  float s = 0.f;
#pragma unroll
  for (int i = 0; i < 8; ++i) { x[i] = __expf(x[i] - m); s += x[i]; }
  for (int off = 16; off > 0; off >>= 1) s += __shfl_down(s, off, 32);
  if ((tid & 31) == 0) red[tid >> 5] = s;
  __syncthreads();
  if (tid < 32) {
    float t = (tid < 8) ? red[tid] : 0.f;
    for (int off = 4; off > 0; off >>= 1) t += __shfl_down(t, off, 32);
    if (tid == 0) red[0] = t;
  }
  __syncthreads();
  float inv = 1.0f / red[0];
#pragma unroll
  for (int i = 0; i < 8; ++i) p[tid + i * 256] = x[i] * inv;
}

// ---------------- ctx[bh,q,d] = attn @ Vh (64 rows x all 64 d per block) ----------------
__global__ __launch_bounds__(256) void av_kernel(const float* __restrict__ attn,
                                                 const unsigned short* __restrict__ Vh,
                                                 unsigned short* __restrict__ Ctx) {
  __shared__ unsigned short At[64 * AST];
  __shared__ unsigned short Bt[64 * AST];
  const int tid = threadIdx.x, lane = tid & 31, wave = tid >> 5;
  const int wm = wave >> 1, wn = wave & 1;
  const int bh = blockIdx.y;
  const int mBase = blockIdx.x * 64;
  const int arow = tid >> 2, aseg = (tid & 3) << 3;
  const int bkrow = tid >> 3, bdseg = (tid & 7) << 3;   // 32 k-rows x 8 d each
  v8f acc0 = {}, acc1 = {};

  for (int k0 = 0; k0 < SEQ; k0 += TK) {
    const float4* asrc = (const float4*)(attn + ((size_t)bh * SEQ + mBase + arow) * SEQ + k0 + aseg);
    if (k0 + TK < SEQ) __builtin_prefetch((const char*)asrc + TK * 4, 0, 1);
    float4 f0 = asrc[0], f1 = asrc[1];
    uint4 pk;
    pk.x = pack2_bf16(f0.x, f0.y); pk.y = pack2_bf16(f0.z, f0.w);
    pk.z = pack2_bf16(f1.x, f1.y); pk.w = pack2_bf16(f1.z, f1.w);
    *(uint4*)(At + arow * AST + aseg) = pk;
    // transpose V tile: Bt[d][k]
    uint4 vv = *(const uint4*)(Vh + ((size_t)bh * SEQ + k0 + bkrow) * DK + bdseg);
    Bt[(bdseg + 0) * AST + bkrow] = (unsigned short)(vv.x & 0xFFFFu);
    Bt[(bdseg + 1) * AST + bkrow] = (unsigned short)(vv.x >> 16);
    Bt[(bdseg + 2) * AST + bkrow] = (unsigned short)(vv.y & 0xFFFFu);
    Bt[(bdseg + 3) * AST + bkrow] = (unsigned short)(vv.y >> 16);
    Bt[(bdseg + 4) * AST + bkrow] = (unsigned short)(vv.z & 0xFFFFu);
    Bt[(bdseg + 5) * AST + bkrow] = (unsigned short)(vv.z >> 16);
    Bt[(bdseg + 6) * AST + bkrow] = (unsigned short)(vv.w & 0xFFFFu);
    Bt[(bdseg + 7) * AST + bkrow] = (unsigned short)(vv.w >> 16);
    __syncthreads();
    v16bf a  = load_A_frag(At, wm * 16, 0, AST);
    v16bf b0 = load_B_frag(Bt, wn * 32, 0, AST);
    v16bf b1 = load_B_frag(Bt, wn * 32 + 16, 0, AST);
    acc0 = wmma_bf16(a, b0, acc0);
    acc1 = wmma_bf16(a, b1, acc1);
    __syncthreads();
  }
  const int mLoc = wm * 16 + ((lane >> 4) & 1) * 8;
  const int dCol0 = wn * 32 + (lane & 15);
  const int dCol1 = dCol0 + 16;
#pragma unroll
  for (int r = 0; r < 8; ++r) {
    size_t rowo = ((size_t)bh * SEQ + mBase + mLoc + r) * DK;
    Ctx[rowo + dCol0] = bf16_bits(acc0[r]);
    Ctx[rowo + dCol1] = bf16_bits(acc1[r]);
  }
}

// ---------------- out-proj + residual: Xres = ctx_merged @ Wo^T + bo + q ----------------
__global__ __launch_bounds__(256) void oproj_kernel(const unsigned short* __restrict__ Ctx,
                                                    const unsigned short* __restrict__ Wo,
                                                    const float* __restrict__ bo,
                                                    const float* __restrict__ residual,
                                                    float* __restrict__ Xres) {
  __shared__ unsigned short At[64 * AST];
  __shared__ unsigned short Bt[64 * AST];
  const int tid = threadIdx.x, lane = tid & 31, wave = tid >> 5;
  const int wm = wave >> 1, wn = wave & 1;
  const int mBase = blockIdx.y * 64, nBase = blockIdx.x * 64;
  const int arow = tid >> 2, aseg = (tid & 3) << 3;
  v8f acc0 = {}, acc1 = {};

  for (int k0 = 0; k0 < D_MODEL; k0 += TK) {
    int m = mBase + arow;
    int bb = m >> 11, s = m & (SEQ - 1);
    int h = k0 >> 6;  // TK=32 slice lies within one head
    async_ld_b128(At + arow * AST + aseg,
                  Ctx + (((size_t)(bb * N_HEADS + h)) * SEQ + s) * DK + (k0 & 63) + aseg);
    async_ld_b128(Bt + arow * AST + aseg,
                  Wo + (size_t)(nBase + arow) * D_MODEL + k0 + aseg);
    wait_async();
    __syncthreads();
    v16bf a  = load_A_frag(At, wm * 16, 0, AST);
    v16bf b0 = load_B_frag(Bt, wn * 32, 0, AST);
    v16bf b1 = load_B_frag(Bt, wn * 32 + 16, 0, AST);
    acc0 = wmma_bf16(a, b0, acc0);
    acc1 = wmma_bf16(a, b1, acc1);
    __syncthreads();
  }
  const int mLoc = wm * 16 + ((lane >> 4) & 1) * 8;
  const int nCol0 = nBase + wn * 32 + (lane & 15);
  const int nCol1 = nCol0 + 16;
  const float bv0 = bo[nCol0], bv1 = bo[nCol1];
#pragma unroll
  for (int r = 0; r < 8; ++r) {
    size_t o = (size_t)(mBase + mLoc + r) * D_MODEL;
    Xres[o + nCol0] = acc0[r] + bv0 + residual[o + nCol0];
    Xres[o + nCol1] = acc1[r] + bv1 + residual[o + nCol1];
  }
}

// ---------------- layernorm rows of 1024 ----------------
__global__ __launch_bounds__(256) void layernorm_kernel(const float* __restrict__ X,
                                                        const float* __restrict__ gamma,
                                                        const float* __restrict__ beta,
                                                        float* __restrict__ Out) {
  __shared__ float red[8];
  const float* p = X + (size_t)blockIdx.x * D_MODEL;
  float* o = Out + (size_t)blockIdx.x * D_MODEL;
  const int tid = threadIdx.x;
  float x[4];
  float s = 0.f;
#pragma unroll
  for (int i = 0; i < 4; ++i) { x[i] = p[tid + i * 256]; s += x[i]; }
  for (int off = 16; off > 0; off >>= 1) s += __shfl_down(s, off, 32);
  if ((tid & 31) == 0) red[tid >> 5] = s;
  __syncthreads();
  if (tid < 32) {
    float t = (tid < 8) ? red[tid] : 0.f;
    for (int off = 4; off > 0; off >>= 1) t += __shfl_down(t, off, 32);
    if (tid == 0) red[0] = t;
  }
  __syncthreads();
  float mu = red[0] * (1.0f / D_MODEL);
  __syncthreads();
  float vs = 0.f;
#pragma unroll
  for (int i = 0; i < 4; ++i) { float d = x[i] - mu; vs += d * d; }
  for (int off = 16; off > 0; off >>= 1) vs += __shfl_down(vs, off, 32);
  if ((tid & 31) == 0) red[tid >> 5] = vs;
  __syncthreads();
  if (tid < 32) {
    float t = (tid < 8) ? red[tid] : 0.f;
    for (int off = 4; off > 0; off >>= 1) t += __shfl_down(t, off, 32);
    if (tid == 0) red[0] = t;
  }
  __syncthreads();
  float rinv = rsqrtf(red[0] * (1.0f / D_MODEL) + LN_EPS);
#pragma unroll
  for (int i = 0; i < 4; ++i) {
    int c = tid + i * 256;
    o[c] = (x[i] - mu) * rinv * gamma[c] + beta[c];
  }
}

extern "C" void kernel_launch(void* const* d_in, const int* in_sizes, int n_in,
                              void* d_out, int out_size, void* d_ws, size_t ws_size,
                              hipStream_t stream) {
  (void)in_sizes; (void)n_in; (void)out_size; (void)ws_size;
  const float* q     = (const float*)d_in[0];
  const float* k     = (const float*)d_in[1];
  const float* v     = (const float*)d_in[2];
  const float* Wq    = (const float*)d_in[3];
  const float* bq    = (const float*)d_in[4];
  const float* Wk    = (const float*)d_in[5];
  const float* bk    = (const float*)d_in[6];
  const float* Wv    = (const float*)d_in[7];
  const float* bv    = (const float*)d_in[8];
  const float* Wo    = (const float*)d_in[9];
  const float* bo    = (const float*)d_in[10];
  const float* gamma = (const float*)d_in[11];
  const float* beta  = (const float*)d_in[12];

  char* ws = (char*)d_ws;
  size_t off = 0;
  const size_t wBytes = (size_t)D_MODEL * D_MODEL * 2;
  const size_t hBytes = (size_t)BATCH * N_HEADS * SEQ * DK * 2;
  unsigned short* Wqb = (unsigned short*)(ws + off); off += wBytes;
  unsigned short* Wkb = (unsigned short*)(ws + off); off += wBytes;
  unsigned short* Wvb = (unsigned short*)(ws + off); off += wBytes;
  unsigned short* Wob = (unsigned short*)(ws + off); off += wBytes;
  unsigned short* Qh  = (unsigned short*)(ws + off); off += hBytes;
  unsigned short* Kh  = (unsigned short*)(ws + off); off += hBytes;
  unsigned short* Vh  = (unsigned short*)(ws + off); off += hBytes;
  unsigned short* Ctx = (unsigned short*)(ws + off); off += hBytes;
  float* Xres = (float*)(ws + off);  // 32 MiB; total ~104 MiB

  float* outp = (float*)d_out;
  float* attn = outp + (size_t)M_TOK * D_MODEL;

  const int nW = D_MODEL * D_MODEL;
  convert_bf16_kernel<<<1024, 256, 0, stream>>>(Wq, Wqb, nW);
  convert_bf16_kernel<<<1024, 256, 0, stream>>>(Wk, Wkb, nW);
  convert_bf16_kernel<<<1024, 256, 0, stream>>>(Wv, Wvb, nW);
  convert_bf16_kernel<<<1024, 256, 0, stream>>>(Wo, Wob, nW);

  dim3 gProj(D_MODEL / 64, M_TOK / 64);              // (16, 128)
  proj_qkv_kernel<<<gProj, 256, 0, stream>>>(q, Wqb, bq, Qh);
  proj_qkv_kernel<<<gProj, 256, 0, stream>>>(k, Wkb, bk, Kh);
  proj_qkv_kernel<<<gProj, 256, 0, stream>>>(v, Wvb, bv, Vh);

  dim3 gScores(SEQ / 64, SEQ / 64, BATCH * N_HEADS); // (32, 32, 64)
  scores_kernel<<<gScores, 256, 0, stream>>>(Qh, Kh, attn);

  softmax_kernel<<<BATCH * N_HEADS * SEQ, 256, 0, stream>>>(attn);

  dim3 gAV(SEQ / 64, BATCH * N_HEADS);               // (32, 64)
  av_kernel<<<gAV, 256, 0, stream>>>(attn, Vh, Ctx);

  oproj_kernel<<<gProj, 256, 0, stream>>>(Ctx, Wob, bo, q, Xres);

  layernorm_kernel<<<M_TOK, 256, 0, stream>>>(Xres, gamma, beta, outp);
}
